// CausalSelfAttention_61220463837806
// MI455X (gfx1250) — matrix operand order repaired and checked by reading, not compiled, not run on hardware
//
#include <hip/hip_runtime.h>
#include <hip/hip_bf16.h>
#include <math.h>

typedef __attribute__((ext_vector_type(16))) _Float16 v16h;
typedef __attribute__((ext_vector_type(8)))  _Float16 h8;
typedef __attribute__((ext_vector_type(8)))  float    v8f;
typedef __attribute__((ext_vector_type(4)))  float    f32x4;
typedef __attribute__((ext_vector_type(4)))  unsigned int u32x4;
typedef __attribute__((ext_vector_type(4)))  int      i32x4;
typedef __attribute__((ext_vector_type(8)))  int      i32x8;

#define NB   2
#define NT   2048
#define NDIM 1024
#define NH   8
#define NHD  128
#define NM   (NB*NT)          /* 4096 rows of x */
#define NQKV (3*NH*NHD)       /* 3072 */
#define ATTN_SCALE 0.12f
#define F32_EPS 1.1920928955078125e-07f

#if __has_include(<hip/amd_detail/amd_gfx1250_TDM.h>)
#define TDM_ARITY6 1
#else
#define TDM_ARITY6 0
#endif

union AFrag { v16h v; struct { h8 lo, hi; } p; };

static __device__ inline v8f wmma16(v16h a, v16h b, v8f c) {
  return __builtin_amdgcn_wmma_f32_16x16x32_f16(false, a, false, b, (short)0, c, false, false);
}

// ---- Tensor Data Mover: 2D f16 tile (tile_d0 contiguous elems x tile_d1 rows,
// ---- row stride stride0 elems) from global -> contiguous LDS at lds_dst.
static __device__ inline void tdm_load_tile_f16(void* lds_dst, const _Float16* g,
                                                unsigned tile_d0, unsigned tile_d1,
                                                unsigned long long stride0) {
  unsigned long long ga = (unsigned long long)(size_t)g;
  unsigned lds = (unsigned)(size_t)lds_dst;
  u32x4 g0;
  g0[0] = 1u;                                       // count=1 (valid), user mode
  g0[1] = lds;                                      // lds_addr
  g0[2] = (unsigned)(ga & 0xFFFFFFFFu);             // global_addr[31:0]
  g0[3] = (unsigned)((ga >> 32) & 0x1FFFFFFu) | (2u << 30);   // addr[56:32] | type=2
  i32x8 g1;
  g1[0] = (int)(1u << 16);                          // data_size=1 (2 bytes), wg_mask=0
  g1[1] = (int)((tile_d0 & 0xFFFFu) << 16);         // tensor_dim0[15:0]
  g1[2] = (int)(((tile_d0 >> 16) & 0xFFFFu) | ((tile_d1 & 0xFFFFu) << 16)); // td0 hi | td1 lo
  g1[3] = (int)(((tile_d1 >> 16) & 0xFFFFu) | ((tile_d0 & 0xFFFFu) << 16)); // td1 hi | tile_dim0
  g1[4] = (int)(tile_d1 & 0xFFFFu);                 // tile_dim1 | tile_dim2=0
  g1[5] = (int)(stride0 & 0xFFFFFFFFull);           // tensor_dim0_stride[31:0]
  g1[6] = (int)((stride0 >> 32) & 0xFFFFull);       // stride[47:32] | dim1_stride lo = 0
  g1[7] = 0;
  i32x4 z4 = {0, 0, 0, 0};
#if TDM_ARITY6
  i32x8 z8 = {0, 0, 0, 0, 0, 0, 0, 0};
  __builtin_amdgcn_tensor_load_to_lds(g0, g1, z4, z4, z8, 0);
#else
  __builtin_amdgcn_tensor_load_to_lds(g0, g1, z4, z4, 0);
#endif
}

static __device__ inline float wred_sum(float x) {
#pragma unroll
  for (int m = 16; m >= 1; m >>= 1) x += __shfl_xor(x, m, 32);
  return x;
}
static __device__ inline float wred_max(float x) {
#pragma unroll
  for (int m = 16; m >= 1; m >>= 1) x = fmaxf(x, __shfl_xor(x, m, 32));
  return x;
}
static __device__ inline float wred_min(float x) {
#pragma unroll
  for (int m = 16; m >= 1; m >>= 1) x = fminf(x, __shfl_xor(x, m, 32));
  return x;
}

// two-sided 8-bit quantization, num_steps = 127, round-half-even (rintf)
static __device__ inline float qlin(float x, float pmax, float nmin) {
  return x >= 0.f ? rintf(x / pmax * 127.f) * (pmax * (1.f / 127.f))
                  : rintf(x / nmin * 127.f) * (nmin * (1.f / 127.f));
}

// ---------------- weight scales: mean(|w|) per 1024x1024 matrix ----------------
__global__ __launch_bounds__(256) void wscale_kernel(const float* __restrict__ w,
                                                     float* __restrict__ scales) {
  __shared__ float red[256];
  int mat = blockIdx.x; // 0..3
  const float* p = w + (size_t)mat * 1024 * 1024;
  float s = 0.f;
  for (int i = threadIdx.x; i < 1024 * 1024; i += 256) s += fabsf(p[i]);
  red[threadIdx.x] = s;
  __syncthreads();
  for (int st = 128; st > 0; st >>= 1) {
    if (threadIdx.x < st) red[threadIdx.x] += red[threadIdx.x + st];
    __syncthreads();
  }
  if (threadIdx.x == 0) scales[mat] = fmaxf(red[0] * (1.f / (1024.f * 1024.f)), 1e-5f);
}

// ---------------- ternary-quantize weights to f16 ----------------
__global__ __launch_bounds__(256) void wquant_kernel(const float* __restrict__ w,
                                                     const float* __restrict__ scales,
                                                     _Float16* __restrict__ wq,
                                                     _Float16* __restrict__ wo) {
  size_t base = ((size_t)blockIdx.x * 256 + threadIdx.x) * 4;
  int mat = (int)(base >> 20);
  float s = scales[mat];
#pragma unroll
  for (int u = 0; u < 4; ++u) {
    size_t i = base + u;
    float q = fminf(fmaxf(rintf(w[i] / s), -1.f), 1.f) * s;
    if (i < (size_t)3 * 1024 * 1024) wq[i] = (_Float16)q;
    else                             wo[i - (size_t)3 * 1024 * 1024] = (_Float16)q;
  }
}

// ---------------- per-row 8-bit quantize of x + gate ----------------
__global__ __launch_bounds__(256) void xquant_kernel(const float* __restrict__ x,
                                                     const float* __restrict__ gw,
                                                     _Float16* __restrict__ xq,
                                                     float* __restrict__ gate) {
  __shared__ float rmin[256], rmax[256];
  __shared__ float sh12[12];
  int row = blockIdx.x;
  const float* xr = x + (size_t)row * NDIM;
  f32x4 xv = ((const f32x4*)xr)[threadIdx.x];
  float mn = fminf(fminf(xv.x, xv.y), fminf(xv.z, xv.w));
  float mx = fmaxf(fmaxf(xv.x, xv.y), fmaxf(xv.z, xv.w));
  rmin[threadIdx.x] = mn; rmax[threadIdx.x] = mx;
  __syncthreads();
  for (int st = 128; st > 0; st >>= 1) {
    if (threadIdx.x < st) {
      rmin[threadIdx.x] = fminf(rmin[threadIdx.x], rmin[threadIdx.x + st]);
      rmax[threadIdx.x] = fmaxf(rmax[threadIdx.x], rmax[threadIdx.x + st]);
    }
    __syncthreads();
  }
  float nmin = fminf(rmin[0], -1e-5f);
  float pmax = fmaxf(rmax[0],  1e-5f);
#pragma unroll
  for (int u = 0; u < 4; ++u) {
    float xe = (u == 0) ? xv.x : (u == 1) ? xv.y : (u == 2) ? xv.z : xv.w;
    float q = qlin(xe, pmax, nmin);
    int col = threadIdx.x * 4 + u;
    xq[(size_t)row * NDIM + col] = (_Float16)q;
    if (col < 12) sh12[col] = q;
  }
  __syncthreads();
  if (threadIdx.x < NH) {
    float z = 0.f;
#pragma unroll
    for (int f = 0; f < 12; ++f) z += sh12[f] * gw[threadIdx.x * 12 + f];
    gate[(size_t)row * NH + threadIdx.x] = 1.f / (1.f + expf(-z));
  }
}

// ---------------- tiled f16 WMMA GEMM: C[M][N] = A[M][K] * B[N][K]^T ----------------
// Tiles staged via the Tensor Data Mover (TENSOR_LOAD_TO_LDS, TENSORcnt).
__global__ __launch_bounds__(256) void gemm_f16_kernel(const _Float16* __restrict__ A,
                                                       const _Float16* __restrict__ Bw,
                                                       float* __restrict__ C,
                                                       int M, int N, int K) {
  __shared__ _Float16 As[64][32];
  __shared__ _Float16 Bs[128][32];
  int m0 = blockIdx.x * 64, n0 = blockIdx.y * 128;
  int w = threadIdx.x >> 5, lane = threadIdx.x & 31;
  int rt = w & 3, cg = w >> 2;
  int mm = lane & 15, hlf = lane >> 4;
  v8f acc[4] = {};
  for (int k0 = 0; k0 < K; k0 += 32) {
    if (threadIdx.x < 32) {
      tdm_load_tile_f16(&As[0][0], A + (size_t)m0 * K + k0, 32, 64, (unsigned long long)K);
      tdm_load_tile_f16(&Bs[0][0], Bw + (size_t)n0 * K + k0, 32, 128, (unsigned long long)K);
      __builtin_amdgcn_s_wait_tensorcnt(0);
    }
    __syncthreads();
    AFrag a, bf;
    const _Float16* ap = &As[rt * 16 + mm][hlf * 8];
    a.p.lo = *(const h8*)(ap);
    a.p.hi = *(const h8*)(ap + 16);
#pragma unroll
    for (int ct = 0; ct < 4; ++ct) {
      const _Float16* bp = &Bs[cg * 64 + ct * 16 + mm][hlf * 8];
      bf.p.lo = *(const h8*)(bp);
      bf.p.hi = *(const h8*)(bp + 16);
      acc[ct] = wmma16(a.v, bf.v, acc[ct]);
    }
    __syncthreads();
  }
  int rb = m0 + rt * 16 + hlf * 8;
  int cb = n0 + cg * 64 + mm;
#pragma unroll
  for (int ct = 0; ct < 4; ++ct)
#pragma unroll
    for (int u = 0; u < 8; ++u)
      C[(size_t)(rb + u) * N + cb + ct * 16] = acc[ct][u];
}

// ---------------- rmsnorm + rotary + requantize q/k + v mix ----------------
__global__ __launch_bounds__(256) void qkvpost_kernel(const float* __restrict__ qkv,
                                                      const float* __restrict__ ve,
                                                      const float* __restrict__ lam,
                                                      const float* __restrict__ cosb,
                                                      const float* __restrict__ sinb,
                                                      _Float16* __restrict__ qf,
                                                      _Float16* __restrict__ kf,
                                                      _Float16* __restrict__ vf) {
  int bt = blockIdx.x;
  int b = bt / NT, t = bt % NT;
  int h = threadIdx.x >> 5;
  int lane = threadIdx.x & 31;
  const float* qr = qkv + (size_t)bt * NQKV + h * NHD;
  const float* kr = qr + NH * NHD;
  const float* vr = qr + 2 * NH * NHD;
  int j = lane * 2;
  float q0 = qr[j], q1 = qr[j + 1], q2 = qr[j + 64], q3 = qr[j + 65];
  float k0 = kr[j], k1 = kr[j + 1], k2 = kr[j + 64], k3 = kr[j + 65];
  float sq = wred_sum(q0 * q0 + q1 * q1 + q2 * q2 + q3 * q3);
  float sk = wred_sum(k0 * k0 + k1 * k1 + k2 * k2 + k3 * k3);
  float rq = rsqrtf(sq * (1.f / 128.f) + F32_EPS);
  float rk = rsqrtf(sk * (1.f / 128.f) + F32_EPS);
  q0 *= rq; q1 *= rq; q2 *= rq; q3 *= rq;
  k0 *= rk; k1 *= rk; k2 *= rk; k3 *= rk;
  float c0 = cosb[t * 64 + j], c1 = cosb[t * 64 + j + 1];
  float s0 = sinb[t * 64 + j], s1 = sinb[t * 64 + j + 1];
  float a0 =  q0 * c0 + q2 * s0, b0 = -q0 * s0 + q2 * c0;
  float a1 =  q1 * c1 + q3 * s1, b1 = -q1 * s1 + q3 * c1;
  float e0 =  k0 * c0 + k2 * s0, f0 = -k0 * s0 + k2 * c0;
  float e1 =  k1 * c1 + k3 * s1, f1 = -k1 * s1 + k3 * c1;
  float qn = fminf(wred_min(fminf(fminf(a0, a1), fminf(b0, b1))), -1e-5f);
  float qp = fmaxf(wred_max(fmaxf(fmaxf(a0, a1), fmaxf(b0, b1))),  1e-5f);
  a0 = qlin(a0, qp, qn); a1 = qlin(a1, qp, qn);
  b0 = qlin(b0, qp, qn); b1 = qlin(b1, qp, qn);
  float kn = fminf(wred_min(fminf(fminf(e0, e1), fminf(f0, f1))), -1e-5f);
  float kp = fmaxf(wred_max(fmaxf(fmaxf(e0, e1), fmaxf(f0, f1))),  1e-5f);
  e0 = qlin(e0, kp, kn); e1 = qlin(e1, kp, kn);
  f0 = qlin(f0, kp, kn); f1 = qlin(f1, kp, kn);
  float l0 = lam[0], l1 = lam[1];
  const float* ver = ve + (size_t)bt * NDIM + h * NHD;
  float v0 = l0 * vr[j]      + l1 * ver[j];
  float v1 = l0 * vr[j + 1]  + l1 * ver[j + 1];
  float v2 = l0 * vr[j + 64] + l1 * ver[j + 64];
  float v3 = l0 * vr[j + 65] + l1 * ver[j + 65];
  size_t base = ((size_t)(b * NH + h) * NT + t) * NHD;
  qf[base + j] = (_Float16)a0; qf[base + j + 1]  = (_Float16)a1;
  qf[base + j + 64] = (_Float16)b0; qf[base + j + 65] = (_Float16)b1;
  kf[base + j] = (_Float16)e0; kf[base + j + 1]  = (_Float16)e1;
  kf[base + j + 64] = (_Float16)f0; kf[base + j + 65] = (_Float16)f1;
  vf[base + j] = (_Float16)v0; vf[base + j + 1]  = (_Float16)v1;
  vf[base + j + 64] = (_Float16)v2; vf[base + j + 65] = (_Float16)v3;
}

// ---------------- flash attention: per (b,h), 64-row q block ----------------
__global__ __launch_bounds__(256) void attn_kernel(const _Float16* __restrict__ qf,
                                                   const _Float16* __restrict__ kf,
                                                   const _Float16* __restrict__ vf,
                                                   const float* __restrict__ gate,
                                                   _Float16* __restrict__ yf) {
  __shared__ _Float16 Qs[64][128];          // 16 KB (TDM-loaded)
  __shared__ _Float16 Xs[64][128];          // 16 KB: K tile (TDM), reused as f32 S [64][64]
  __shared__ _Float16 Ps[64][64];           // 8 KB: exp(S-m) in f16
  __shared__ _Float16 Vt[128][64];          // 16 KB: V^T [hd][key] (manual transpose)
  __shared__ float mrow[64], lrow[64], arow[64];
  float* Sf = (float*)&Xs[0][0];            // ld = 64 floats

  int qb = blockIdx.x;
  int bh = blockIdx.y;
  int b = bh >> 3, h = bh & 7;
  int q0 = qb * 64;
  const _Float16* qbase = qf + (size_t)bh * NT * NHD;
  const _Float16* kbase = kf + (size_t)bh * NT * NHD;
  const _Float16* vbase = vf + (size_t)bh * NT * NHD;
  int w = threadIdx.x >> 5, lane = threadIdx.x & 31;
  int rt = w & 3, cg = w >> 2;
  int mm = lane & 15, hlf = lane >> 4;

  if (threadIdx.x < 32) {
    tdm_load_tile_f16(&Qs[0][0], qbase + (size_t)q0 * NHD, 128, 64, 128ull);
    __builtin_amdgcn_s_wait_tensorcnt(0);
  }
  if (threadIdx.x < 64) { mrow[threadIdx.x] = -INFINITY; lrow[threadIdx.x] = 0.f; }
  v8f ya[4] = {};
  __syncthreads();

  for (int kb = 0; kb <= qb; ++kb) {
    int kg = kb * 64;
    if (threadIdx.x < 32)           // async DMA of K tile, overlapped with V transpose
      tdm_load_tile_f16(&Xs[0][0], kbase + (size_t)kg * NHD, 128, 64, 128ull);
    for (int i = threadIdx.x; i < 64 * 16; i += 256) {
      int r = i >> 4, c = (i & 15) * 8;
      union { u32x4 u; _Float16 hh[8]; } cv;
      cv.u = *(const u32x4*)(vbase + (size_t)(kg + r) * NHD + c);
#pragma unroll
      for (int u = 0; u < 8; ++u) Vt[c + u][r] = cv.hh[u];
    }
    if (threadIdx.x < 32) __builtin_amdgcn_s_wait_tensorcnt(0);
    __syncthreads();
    // S = Q * K^T (this wave: 16 rows x 32 cols)
    v8f sacc[2] = {};
    AFrag a, bf;
#pragma unroll
    for (int kk = 0; kk < 128; kk += 32) {
      const _Float16* ap = &Qs[rt * 16 + mm][kk + hlf * 8];
      a.p.lo = *(const h8*)(ap);
      a.p.hi = *(const h8*)(ap + 16);
#pragma unroll
      for (int ci = 0; ci < 2; ++ci) {
        const _Float16* bp = &Xs[cg * 32 + ci * 16 + mm][kk + hlf * 8];
        bf.p.lo = *(const h8*)(bp);
        bf.p.hi = *(const h8*)(bp + 16);
        sacc[ci] = wmma16(a.v, bf.v, sacc[ci]);
      }
    }
    __syncthreads();                 // all waves done reading K tile
    {
      int rb = rt * 16 + hlf * 8;
      int cb = cg * 32 + mm;
#pragma unroll
      for (int ci = 0; ci < 2; ++ci)
#pragma unroll
        for (int u = 0; u < 8; ++u)
          Sf[(rb + u) * 64 + cb + ci * 16] = sacc[ci][u] * ATTN_SCALE;
    }
    __syncthreads();
    // online softmax per row (threads 0..63 each own a row)
    if (threadIdx.x < 64) {
      int r = threadIdx.x;
      int vc = q0 + r - kg + 1; if (vc > 64) vc = 64;   // causal valid cols
      float mx = -3.0e38f;
      for (int c = 0; c < vc; ++c) mx = fmaxf(mx, Sf[r * 64 + c]);
      float mo = mrow[r], mn = fmaxf(mo, mx);
      float al = __expf(mo - mn);
      float sum = 0.f;
      for (int c = 0; c < 64; ++c) {
        float p = (c < vc) ? __expf(Sf[r * 64 + c] - mn) : 0.f;
        sum += p;
        Ps[r][c] = (_Float16)p;
      }
      lrow[r] = lrow[r] * al + sum;
      mrow[r] = mn;
      arow[r] = al;
    }
    __syncthreads();
    // rescale accumulators, then y += P * V
#pragma unroll
    for (int ct = 0; ct < 4; ++ct)
#pragma unroll
      for (int u = 0; u < 8; ++u)
        ya[ct][u] *= arow[rt * 16 + hlf * 8 + u];
#pragma unroll
    for (int kk = 0; kk < 64; kk += 32) {
      const _Float16* ap = &Ps[rt * 16 + mm][kk + hlf * 8];
      a.p.lo = *(const h8*)(ap);
      a.p.hi = *(const h8*)(ap + 16);
#pragma unroll
      for (int ct = 0; ct < 4; ++ct) {
        const _Float16* bp = &Vt[cg * 64 + ct * 16 + mm][kk + hlf * 8];
        bf.p.lo = *(const h8*)(bp);
        bf.p.hi = *(const h8*)(bp + 16);
        ya[ct] = wmma16(a.v, bf.v, ya[ct]);
      }
    }
    __syncthreads();                 // before next iteration reuses Xs/Vt
  }
  // epilogue: y / l * gate -> f16 [b][t][h*128+hd]
#pragma unroll
  for (int ct = 0; ct < 4; ++ct) {
    int colg = cg * 64 + ct * 16 + mm;
#pragma unroll
    for (int u = 0; u < 8; ++u) {
      int r = rt * 16 + hlf * 8 + u;
      int tg = q0 + r;
      float val = ya[ct][u] / lrow[r] * gate[((size_t)b * NT + tg) * NH + h];
      yf[((size_t)b * NT + tg) * (NH * NHD) + h * NHD + colg] = (_Float16)val;
    }
  }
}

extern "C" void kernel_launch(void* const* d_in, const int* in_sizes, int n_in,
                              void* d_out, int out_size, void* d_ws, size_t ws_size,
                              hipStream_t stream) {
  const float* x    = (const float*)d_in[0];
  const float* ve   = (const float*)d_in[1];
  const float* lam  = (const float*)d_in[2];
  const float* cosb = (const float*)d_in[3];
  const float* sinb = (const float*)d_in[4];
  const float* w    = (const float*)d_in[5];
  const float* gw   = (const float*)d_in[6];
  float* out = (float*)d_out;

  char* ws = (char*)d_ws;
  size_t off = 0;
  auto alloc = [&](size_t bytes) {
    size_t r = off;
    off = (off + bytes + 255) & ~(size_t)255;
    return r;
  };
  float*    scales = (float*)   (ws + alloc(4 * sizeof(float)));
  _Float16* xq     = (_Float16*)(ws + alloc((size_t)NM * NDIM * 2));
  float*    gate   = (float*)   (ws + alloc((size_t)NM * NH * 4));
  _Float16* wq     = (_Float16*)(ws + alloc((size_t)NQKV * NDIM * 2));
  _Float16* wo     = (_Float16*)(ws + alloc((size_t)NDIM * NDIM * 2));
  float*    qkv    = (float*)   (ws + alloc((size_t)NM * NQKV * 4));
  _Float16* qf     = (_Float16*)(ws + alloc((size_t)NB * NH * NT * NHD * 2));
  _Float16* kf     = (_Float16*)(ws + alloc((size_t)NB * NH * NT * NHD * 2));
  _Float16* vf     = (_Float16*)(ws + alloc((size_t)NB * NH * NT * NHD * 2));
  _Float16* yf     = (_Float16*)(ws + alloc((size_t)NM * NDIM * 2));

  wscale_kernel<<<4, 256, 0, stream>>>(w, scales);
  wquant_kernel<<<4096, 256, 0, stream>>>(w, scales, wq, wo);
  xquant_kernel<<<NM, 256, 0, stream>>>(x, gw, xq, gate);
  gemm_f16_kernel<<<dim3(NM / 64, NQKV / 128), 256, 0, stream>>>(xq, wq, qkv, NM, NQKV, NDIM);
  qkvpost_kernel<<<NM, 256, 0, stream>>>(qkv, ve, lam, cosb, sinb, qf, kf, vf);
  attn_kernel<<<dim3(NT / 64, NB * NH), 256, 0, stream>>>(qf, kf, vf, gate, yf);
  gemm_f16_kernel<<<dim3(NM / 64, NDIM / 128), 256, 0, stream>>>(yf, wo, out, NM, NDIM, NDIM);
}
